// myModel_584115552485
// MI455X (gfx1250) — matrix operand order repaired and checked by reading, not compiled, not run on hardware
//
#include <hip/hip_runtime.h>
#include <cstdint>
#include <cstddef>

// ---------------- model dims ----------------
#define BB 8
#define S_TOT 144
#define MAX_STMT 128
#define TT 20
#define DD 768
#define VHD 192
#define NC 150
#define NH 12
#define DH 64
#define FF 3072
#define LSEQ 129           // MAX_STMT + 1
#define M_T5 1056          // 8*129 = 1032 padded to %32
#define M_T5_REAL 1032
#define D3 2304            // 3*D
#define MPAD 32            // padded row count for batch-8 recurrent states

typedef unsigned short u16;
typedef __attribute__((ext_vector_type(16))) __bf16 v16bf;
typedef __attribute__((ext_vector_type(8)))  float  v8f;

union FragBF { v16bf v; uint4 q[2]; };

__device__ __forceinline__ u16 f32_to_bf16(float f) {
  unsigned u = __float_as_uint(f);
  u += 0x7FFFu + ((u >> 16) & 1u);          // round-to-nearest-even
  return (u16)(u >> 16);
}
__device__ __forceinline__ float sigmoidf(float x) { return 1.0f / (1.0f + __expf(-x)); }

// ---------------- tiny utility kernels ----------------
__global__ void k_cvt_f2b(const float* __restrict__ in, u16* __restrict__ out, size_t n) {
  size_t i = (size_t)blockIdx.x * blockDim.x + threadIdx.x;
  if (i < n) out[i] = f32_to_bf16(in[i]);
}
__global__ void k_zero_f32(float* p, size_t n) {
  size_t i = (size_t)blockIdx.x * blockDim.x + threadIdx.x;
  if (i < n) p[i] = 0.0f;
}
__global__ void k_zero_u16(u16* p, size_t n) {
  size_t i = (size_t)blockIdx.x * blockDim.x + threadIdx.x;
  if (i < n) p[i] = 0;
}

// embed gather: X[token, d] = bf16(embed[ids[token], d])
__global__ void k_embed(const int* __restrict__ ids, const float* __restrict__ embed,
                        u16* __restrict__ X, int ntok) {
  size_t i = (size_t)blockIdx.x * blockDim.x + threadIdx.x;
  if (i >= (size_t)ntok * DD) return;
  int t = (int)(i / DD), d = (int)(i % DD);
  X[i] = f32_to_bf16(embed[(size_t)ids[t] * DD + d]);
}

// ---------------- WMMA GEMM: C[M,N] = A[M,K] * W[N,K]^T (+bias,+residual,act) ----------
// A,W bf16 (K-contig). Each wave: 32M x 64N tile, 2 A-frags reused over 4 W-frags,
// 8 accumulators, 8 v_wmma per K-step of 32. Requires M%32==0, N%64==0, K%32==0.
#define ACT_NONE 0
#define ACT_RELU 1
#define ACT_TANH 2
__global__ __launch_bounds__(256)
void k_gemm_bf16(const u16* __restrict__ A, const u16* __restrict__ W,
                 const float* __restrict__ bias, const float* __restrict__ residual,
                 float* __restrict__ outF, u16* __restrict__ outBF,
                 int M, int N, int K, int act) {
  const int lane = threadIdx.x & 31;
  const int wave = threadIdx.x >> 5;
  const int nt = N >> 6;                       // 64-wide N tiles
  const int mt = M >> 5;                       // 32-tall M tiles
  int gw = blockIdx.x * 8 + wave;
  if (gw >= mt * nt) return;
  const int m0 = (gw / nt) << 5;
  const int n0 = (gw % nt) << 6;

  const int hi = lane >> 4;                    // half-wave (K-split per ISA layout)
  const int lr = lane & 15;

  v8f acc[8] = {};
  const size_t arow0 = (size_t)(m0 + lr) * K;
  const size_t arow1 = arow0 + (size_t)16 * K;
  const size_t wrow  = (size_t)(n0 + lr) * K;
  const size_t wstep = (size_t)16 * K;

  for (int kk = 0; kk < K; kk += 32) {
    FragBF a0, a1;
    const u16* ap0 = A + arow0 + kk + (hi << 3);   // A frag: chunks kk+{0|8}, kk+16+{0|8}
    const u16* ap1 = A + arow1 + kk + (hi << 3);
    a0.q[0] = *(const uint4*)(ap0); a0.q[1] = *(const uint4*)(ap0 + 16);
    a1.q[0] = *(const uint4*)(ap1); a1.q[1] = *(const uint4*)(ap1 + 16);
    __builtin_prefetch(ap0 + 128, 0, 1);           // next K-tiles -> global_prefetch_b8
    __builtin_prefetch(ap1 + 128, 0, 1);
    const int kb = kk + (hi << 3);                 // B frag: chunks kk+{0|8}, kk+16+{0|8}
#pragma unroll
    for (int j = 0; j < 4; ++j) {
      FragBF b;
      const u16* wp = W + wrow + wstep * j + kb;
      b.q[0] = *(const uint4*)(wp);
      b.q[1] = *(const uint4*)(wp + 16);
      acc[j] = __builtin_amdgcn_wmma_f32_16x16x32_bf16(
          false, a0.v, false, b.v, (short)0, acc[j], false, false);
      acc[j + 4] = __builtin_amdgcn_wmma_f32_16x16x32_bf16(
          false, a1.v, false, b.v, (short)0, acc[j + 4], false, false);
    }
  }
#pragma unroll
  for (int half = 0; half < 2; ++half) {
#pragma unroll
    for (int j = 0; j < 4; ++j) {
      int n = n0 + j * 16 + lr;
      float bs = bias ? bias[n] : 0.0f;
#pragma unroll
      for (int r = 0; r < 8; ++r) {
        int m = m0 + half * 16 + (hi << 3) + r;
        float val = acc[half * 4 + j][r] + bs;
        size_t o = (size_t)m * N + n;
        if (residual) val += residual[o];
        if (act == ACT_RELU)      val = fmaxf(val, 0.0f);
        else if (act == ACT_TANH) val = tanhf(val);
        if (outF)  outF[o]  = val;
        if (outBF) outBF[o] = f32_to_bf16(val);
      }
    }
  }
}

// ---------------- GRU gate combine (xi/gh already include biases) ----------------
__global__ void k_gru_combine(const float* __restrict__ xiBase, int T, int t,
                              const float* __restrict__ gh,
                              float* __restrict__ h, u16* __restrict__ hbf, int Nseq) {
  size_t i = (size_t)blockIdx.x * blockDim.x + threadIdx.x;
  if (i >= (size_t)Nseq * DD) return;
  int seq = (int)(i / DD), d = (int)(i % DD);
  const float* xr = xiBase + (size_t)(seq * T + t) * D3;
  const float* gr = gh + (size_t)seq * D3;
  float r = sigmoidf(xr[d] + gr[d]);
  float z = sigmoidf(xr[DD + d] + gr[DD + d]);
  float n = tanhf(xr[2 * DD + d] + r * gr[2 * DD + d]);
  float hv = (1.0f - z) * n + z * h[i];
  h[i] = hv;
  hbf[i] = f32_to_bf16(hv);
}

// row gather: dst row r <- src row (r/innerN)*outerStride + innerOff + r%innerN
__global__ void k_gather_rows_f2b(const float* __restrict__ src, u16* __restrict__ dst,
                                  int innerN, int outerStride, int innerOff, int nrows, int D) {
  size_t i = (size_t)blockIdx.x * blockDim.x + threadIdx.x;
  if (i >= (size_t)nrows * D) return;
  int r = (int)(i / D), d = (int)(i % D);
  int sr = (r / innerN) * outerStride + innerOff + (r % innerN);
  dst[i] = f32_to_bf16(src[(size_t)sr * D + d]);
}
__global__ void k_gather_rows_b2b(const u16* __restrict__ src, u16* __restrict__ dst,
                                  int innerN, int outerStride, int innerOff, int nrows, int D) {
  size_t i = (size_t)blockIdx.x * blockDim.x + threadIdx.x;
  if (i >= (size_t)nrows * D) return;
  int r = (int)(i / D), d = (int)(i % D);
  int sr = (r / innerN) * outerStride + innerOff + (r % innerN);
  dst[i] = src[(size_t)sr * D + d];
}

// LayerNorm over VH=192 for rows 0..7
__global__ __launch_bounds__(256)
void k_layernorm_v(const float* __restrict__ v, const float* __restrict__ g,
                   const float* __restrict__ b, float* __restrict__ out) {
  int row = blockIdx.x;
  __shared__ float red[256];
  const float* vr = v + (size_t)row * VHD;
  float s = 0.0f;
  for (int d = threadIdx.x; d < VHD; d += 256) s += vr[d];
  red[threadIdx.x] = s; __syncthreads();
  for (int st = 128; st > 0; st >>= 1) { if (threadIdx.x < st) red[threadIdx.x] += red[threadIdx.x + st]; __syncthreads(); }
  float mu = red[0] / VHD; __syncthreads();
  s = 0.0f;
  for (int d = threadIdx.x; d < VHD; d += 256) { float t = vr[d] - mu; s += t * t; }
  red[threadIdx.x] = s; __syncthreads();
  for (int st = 128; st > 0; st >>= 1) { if (threadIdx.x < st) red[threadIdx.x] += red[threadIdx.x + st]; __syncthreads(); }
  float inv = rsqrtf(red[0] / VHD + 1e-5f);
  for (int d = threadIdx.x; d < VHD; d += 256)
    out[(size_t)row * VHD + d] = (vr[d] - mu) * inv * g[d] + b[d];
}

// nearest center (argmin over 150) -> write chosen center (bf16) into row b
__global__ __launch_bounds__(256)
void k_nearest(const float* __restrict__ vn, const float* __restrict__ centers,
               u16* __restrict__ closest_bf) {
  int b = blockIdx.x;
  __shared__ float bd[256]; __shared__ int bi[256];
  float best = 1e30f; int besti = 0;
  for (int c = threadIdx.x; c < NC; c += 256) {
    float s = 0.0f;
    for (int d = 0; d < VHD; ++d) { float t = vn[(size_t)b * VHD + d] - centers[(size_t)c * VHD + d]; s += t * t; }
    if (s < best) { best = s; besti = c; }
  }
  bd[threadIdx.x] = best; bi[threadIdx.x] = besti; __syncthreads();
  for (int st = 128; st > 0; st >>= 1) {
    if (threadIdx.x < st && bd[threadIdx.x + st] < bd[threadIdx.x]) {
      bd[threadIdx.x] = bd[threadIdx.x + st]; bi[threadIdx.x] = bi[threadIdx.x + st];
    }
    __syncthreads();
  }
  int c = bi[0];
  for (int d = threadIdx.x; d < VHD; d += 256)
    closest_bf[(size_t)b * VHD + d] = f32_to_bf16(centers[(size_t)c * VHD + d]);
}

// build T5 input x (M_T5,768): first 128 rows/b from h_stmt, row 128 = center_hi, pad rows zero
__global__ void k_build_inp(const float* __restrict__ hstmt, const float* __restrict__ chi,
                            float* __restrict__ x) {
  size_t i = (size_t)blockIdx.x * blockDim.x + threadIdx.x;
  if (i >= (size_t)M_T5 * DD) return;
  int r = (int)(i / DD), d = (int)(i % DD);
  float v = 0.0f;
  if (r < M_T5_REAL) {
    int b = r / LSEQ, s = r % LSEQ;
    v = (s < MAX_STMT) ? hstmt[(size_t)(b * S_TOT + s) * DD + d]
                       : chi[(size_t)b * DD + d];
  }
  x[i] = v;
}

// T5 relative-position bias (NH,129,129)
__global__ void k_rel_bias(const float* __restrict__ rel, float* __restrict__ bias) {
  int idx = blockIdx.x * blockDim.x + threadIdx.x;
  if (idx >= LSEQ * LSEQ) return;
  int q = idx / LSEQ, k = idx % LSEQ;
  int r = k - q;
  int ret = (r > 0) ? 16 : 0;
  int n = (r < 0) ? -r : r;
  int large = 8 + (int)(logf((float)n / 8.0f + 1e-9f) * (1.0f / logf(16.0f)) * 8.0f);
  if (large > 15) large = 15;
  int bucket = ret + ((n < 8) ? n : large);
  for (int h = 0; h < NH; ++h)
    bias[((size_t)h * LSEQ + q) * LSEQ + k] = rel[bucket * NH + h];
}

// RMSNorm row kernel: out_bf (and optional out_f32)
__global__ __launch_bounds__(256)
void k_rms(const float* __restrict__ x, const float* __restrict__ w,
           u16* __restrict__ obf, float* __restrict__ of32, int D) {
  int row = blockIdx.x;
  __shared__ float red[256];
  const float* xr = x + (size_t)row * D;
  float s = 0.0f;
  for (int d = threadIdx.x; d < D; d += 256) { float v = xr[d]; s += v * v; }
  red[threadIdx.x] = s; __syncthreads();
  for (int st = 128; st > 0; st >>= 1) { if (threadIdx.x < st) red[threadIdx.x] += red[threadIdx.x + st]; __syncthreads(); }
  float scale = rsqrtf(red[0] / D + 1e-6f);
  for (int d = threadIdx.x; d < D; d += 256) {
    float v = xr[d] * scale * w[d];
    obf[(size_t)row * D + d] = f32_to_bf16(v);
    if (of32) of32[(size_t)row * D + d] = v;
  }
}

// fused attention: one block per (q, head, batch); scores + softmax + ctx
__global__ __launch_bounds__(128)
void k_attention(const float* __restrict__ Q, const float* __restrict__ Km,
                 const float* __restrict__ Vm, const float* __restrict__ bias,
                 const float* __restrict__ mask, u16* __restrict__ ctxBF) {
  int qi = blockIdx.x, h = blockIdx.y, b = blockIdx.z;
  __shared__ float sc[LSEQ];
  __shared__ float red[128];
  const float* qrow = Q + (size_t)(b * LSEQ + qi) * DD + h * DH;
  for (int ki = threadIdx.x; ki < LSEQ; ki += 128) {
    const float* krow = Km + (size_t)(b * LSEQ + ki) * DD + h * DH;
    float s = 0.0f;
    for (int d = 0; d < DH; ++d) s += qrow[d] * krow[d];
    s += bias[((size_t)h * LSEQ + qi) * LSEQ + ki];
    s += (1.0f - mask[b * LSEQ + ki]) * -1e9f;
    sc[ki] = s;
  }
  __syncthreads();
  float m = -1e30f;
  for (int ki = threadIdx.x; ki < LSEQ; ki += 128) m = fmaxf(m, sc[ki]);
  red[threadIdx.x] = m; __syncthreads();
  for (int st = 64; st > 0; st >>= 1) { if (threadIdx.x < st) red[threadIdx.x] = fmaxf(red[threadIdx.x], red[threadIdx.x + st]); __syncthreads(); }
  m = red[0]; __syncthreads();
  float sum = 0.0f;
  for (int ki = threadIdx.x; ki < LSEQ; ki += 128) { float e = __expf(sc[ki] - m); sc[ki] = e; sum += e; }
  red[threadIdx.x] = sum; __syncthreads();
  for (int st = 64; st > 0; st >>= 1) { if (threadIdx.x < st) red[threadIdx.x] += red[threadIdx.x + st]; __syncthreads(); }
  float inv = 1.0f / red[0];
  __syncthreads();
  if (threadIdx.x < DH) {
    int d = threadIdx.x;
    float a = 0.0f;
    for (int ki = 0; ki < LSEQ; ++ki) a += sc[ki] * Vm[(size_t)(b * LSEQ + ki) * DD + h * DH + d];
    ctxBF[(size_t)(b * LSEQ + qi) * DD + h * DH + d] = f32_to_bf16(a * inv);
  }
}

// out[row*C + c] = dot(X[row,:D], W[c,:D]) + bias[c] ; grid (rows, C)
__global__ __launch_bounds__(256)
void k_rowdot(const float* __restrict__ X, const float* __restrict__ W,
              const float* __restrict__ bias, float* __restrict__ out, int D, int C) {
  int row = blockIdx.x, c = blockIdx.y;
  __shared__ float red[256];
  float s = 0.0f;
  for (int d = threadIdx.x; d < D; d += 256) s += X[(size_t)row * D + d] * W[(size_t)c * D + d];
  red[threadIdx.x] = s; __syncthreads();
  for (int st = 128; st > 0; st >>= 1) { if (threadIdx.x < st) red[threadIdx.x] += red[threadIdx.x + st]; __syncthreads(); }
  if (threadIdx.x == 0) out[(size_t)row * C + c] = red[0] + bias[c];
}

__global__ __launch_bounds__(256)
void k_stmt_loss(const float* __restrict__ logits, const float* __restrict__ y,
                 float* __restrict__ out) {
  __shared__ float red[256];
  float s = 0.0f;
  for (int i = threadIdx.x; i < BB * MAX_STMT; i += 256) {
    float l = logits[i];
    s += fmaxf(l, 0.0f) - l * y[i] + log1pf(__expf(-fabsf(l)));
  }
  red[threadIdx.x] = s; __syncthreads();
  for (int st = 128; st > 0; st >>= 1) { if (threadIdx.x < st) red[threadIdx.x] += red[threadIdx.x + st]; __syncthreads(); }
  if (threadIdx.x == 0) out[0] = red[0] / (float)(BB * MAX_STMT);
}

__global__ void k_func_loss(const float* __restrict__ fl, const int* __restrict__ labels,
                            float* __restrict__ out) {
  if (threadIdx.x == 0 && blockIdx.x == 0) {
    float s = 0.0f;
    for (int b = 0; b < BB; ++b) {
      float a = fl[b * 2 + 0], c = fl[b * 2 + 1];
      float m = fmaxf(a, c);
      float lse = m + logf(__expf(a - m) + __expf(c - m));
      s += -(fl[b * 2 + labels[b]] - lse);
    }
    out[1] = s / (float)BB;
  }
}

// ---------------- host orchestration ----------------
enum {
  IN_IDS = 0, IN_MASK, IN_YSTMT, IN_LABELS,
  IN_CENTERS,
  IN_CLS_DENSE_B, IN_CLS_DENSE_W, IN_CLS_FDENSE_B, IN_CLS_FDENSE_W,
  IN_CLS_FOUT_B, IN_CLS_FOUT_W,
  IN_CLS_GRU_WH, IN_CLS_GRU_WI, IN_CLS_GRU_BH, IN_CLS_GRU_BI,
  IN_CLS_OUT_B, IN_CLS_OUT_W,
  IN_EMBED,
  IN_GS_WH, IN_GS_WI, IN_GS_BH, IN_GS_BI,
  IN_GV_WH, IN_GV_WI, IN_GV_BH, IN_GV_BI,
  IN_HIGH_B, IN_HIGH_W, IN_LN_B, IN_LN_G,
  IN_LOW_B, IN_LOW_W, IN_REL, IN_T5_FINAL_LN,
  IN_L0_K, IN_L0_LN1, IN_L0_LN2, IN_L0_O, IN_L0_Q, IN_L0_V, IN_L0_WI, IN_L0_WO,
  IN_L1_K, IN_L1_LN1, IN_L1_LN2, IN_L1_O, IN_L1_Q, IN_L1_V, IN_L1_WI, IN_L1_WO
};

static inline int cdiv(size_t a, int b) { return (int)((a + b - 1) / b); }
static inline char* bump(char*& p, size_t bytes) {
  char* r = p; p += (bytes + 255) & ~(size_t)255; return r;
}

extern "C" void kernel_launch(void* const* d_in, const int* in_sizes, int n_in,
                              void* d_out, int out_size, void* d_ws, size_t ws_size,
                              hipStream_t stream) {
  (void)in_sizes; (void)n_in; (void)out_size; (void)ws_size;
  auto fin = [&](int i) { return (const float*)d_in[i]; };

  char* p = (char*)d_ws;
  auto B16 = [&](size_t e) { return (u16*)bump(p, e * 2); };
  auto F32 = [&](size_t e) { return (float*)bump(p, e * 4); };

  // bf16 weight buffers
  u16 *wiS = B16((size_t)D3 * DD), *whS = B16((size_t)D3 * DD);
  u16 *wiV = B16((size_t)D3 * DD), *whV = B16((size_t)D3 * DD);
  u16 *wiC = B16((size_t)D3 * DD), *whC = B16((size_t)D3 * DD);
  u16 *lowW = B16((size_t)VHD * DD), *highW = B16((size_t)DD * VHD);
  u16 *denseW = B16((size_t)DD * DD), *fdenseW = B16((size_t)DD * DD);
  u16 *qb[2], *kb[2], *vb[2], *ob[2], *wib[2], *wob[2];
  for (int l = 0; l < 2; ++l) {
    qb[l] = B16((size_t)DD * DD); kb[l] = B16((size_t)DD * DD);
    vb[l] = B16((size_t)DD * DD); ob[l] = B16((size_t)DD * DD);
    wib[l] = B16((size_t)FF * DD); wob[l] = B16((size_t)DD * FF);
  }
  // bf16 activation buffers
  u16 *Xemb  = B16((size_t)BB * S_TOT * TT * DD);
  u16 *hS_bf = B16((size_t)BB * S_TOT * DD);
  u16 *Avul  = B16((size_t)BB * 16 * DD);
  u16 *hV_bf = B16((size_t)MPAD * DD);
  u16 *clo_bf = B16((size_t)MPAD * VHD);
  u16 *h_bf  = B16((size_t)M_T5 * DD);
  u16 *ctx_bf = B16((size_t)M_T5 * DD);
  u16 *ff_bf = B16((size_t)M_T5 * FF);
  u16 *rep_bf = B16((size_t)M_T5 * DD);
  u16 *hf_bf = B16((size_t)BB * MAX_STMT * DD);
  u16 *hC_bf = B16((size_t)MPAD * DD);
  // f32 scratch
  float *xiS = F32((size_t)BB * S_TOT * TT * D3);
  float *gh  = F32((size_t)BB * S_TOT * D3);
  float *hS  = F32((size_t)BB * S_TOT * DD);
  float *xiV = F32((size_t)BB * 16 * D3);
  float *hV  = F32((size_t)MPAD * DD);
  float *vlo = F32((size_t)MPAD * VHD);
  float *vn  = F32((size_t)BB * VHD);
  float *chi = F32((size_t)MPAD * DD);
  float *xT  = F32((size_t)M_T5 * DD);
  float *qF  = F32((size_t)M_T5 * DD);
  float *kF  = F32((size_t)M_T5 * DD);
  float *vF  = F32((size_t)M_T5 * DD);
  float *biasT = F32((size_t)NH * LSEQ * LSEQ);
  float *xiC = F32((size_t)BB * MAX_STMT * D3);
  float *hC  = F32((size_t)MPAD * DD);
  float *xcls = F32((size_t)BB * MAX_STMT * DD);
  float *logits = F32((size_t)BB * MAX_STMT);
  float *fh  = F32((size_t)MPAD * DD);
  float *flog = F32(16);

  auto cvt = [&](int idx, u16* dst, size_t n) {
    k_cvt_f2b<<<cdiv(n, 256), 256, 0, stream>>>(fin(idx), dst, n);
  };
  auto zf = [&](float* ptr, size_t n) { k_zero_f32<<<cdiv(n, 256), 256, 0, stream>>>(ptr, n); };
  auto zb = [&](u16* ptr, size_t n) { k_zero_u16<<<cdiv(n, 256), 256, 0, stream>>>(ptr, n); };
  auto gemm = [&](const u16* A, const u16* W, const float* bias, const float* res,
                  float* oF, u16* oBF, int M, int N, int K, int act) {
    int waves = (M / 32) * (N / 64);
    k_gemm_bf16<<<cdiv(waves, 8), 256, 0, stream>>>(A, W, bias, res, oF, oBF, M, N, K, act);
  };

  // --- weight conversion (every call; no caching) ---
  cvt(IN_GS_WI, wiS, (size_t)D3 * DD); cvt(IN_GS_WH, whS, (size_t)D3 * DD);
  cvt(IN_GV_WI, wiV, (size_t)D3 * DD); cvt(IN_GV_WH, whV, (size_t)D3 * DD);
  cvt(IN_CLS_GRU_WI, wiC, (size_t)D3 * DD); cvt(IN_CLS_GRU_WH, whC, (size_t)D3 * DD);
  cvt(IN_LOW_W, lowW, (size_t)VHD * DD); cvt(IN_HIGH_W, highW, (size_t)DD * VHD);
  cvt(IN_CLS_DENSE_W, denseW, (size_t)DD * DD); cvt(IN_CLS_FDENSE_W, fdenseW, (size_t)DD * DD);
  int lq[2] = {IN_L0_Q, IN_L1_Q}, lk[2] = {IN_L0_K, IN_L1_K}, lv[2] = {IN_L0_V, IN_L1_V};
  int lo[2] = {IN_L0_O, IN_L1_O}, lwi[2] = {IN_L0_WI, IN_L1_WI}, lwo[2] = {IN_L0_WO, IN_L1_WO};
  int lln1[2] = {IN_L0_LN1, IN_L1_LN1}, lln2[2] = {IN_L0_LN2, IN_L1_LN2};
  for (int l = 0; l < 2; ++l) {
    cvt(lq[l], qb[l], (size_t)DD * DD); cvt(lk[l], kb[l], (size_t)DD * DD);
    cvt(lv[l], vb[l], (size_t)DD * DD); cvt(lo[l], ob[l], (size_t)DD * DD);
    cvt(lwi[l], wib[l], (size_t)FF * DD); cvt(lwo[l], wob[l], (size_t)DD * FF);
  }

  // --- embedding gather -> bf16 tokens ---
  const int NTOK = BB * S_TOT * TT;  // 23040
  k_embed<<<cdiv((size_t)NTOK * DD, 256), 256, 0, stream>>>(
      (const int*)d_in[IN_IDS], fin(IN_EMBED), Xemb, NTOK);

  // --- statement GRU: xi = X @ Wi^T + bi, then 20 recurrent steps ---
  gemm(Xemb, wiS, fin(IN_GS_BI), nullptr, xiS, nullptr, NTOK, D3, DD, ACT_NONE);
  const int NSEQ = BB * S_TOT;  // 1152
  zf(hS, (size_t)NSEQ * DD); zb(hS_bf, (size_t)NSEQ * DD);
  for (int t = 0; t < TT; ++t) {
    gemm(hS_bf, whS, fin(IN_GS_BH), nullptr, gh, nullptr, NSEQ, D3, DD, ACT_NONE);
    k_gru_combine<<<cdiv((size_t)NSEQ * DD, 256), 256, 0, stream>>>(xiS, TT, t, gh, hS, hS_bf, NSEQ);
  }

  // --- vul GRU over stmt_emb[:,128:,:] (8 seqs, 16 steps) ---
  k_gather_rows_f2b<<<cdiv((size_t)BB * 16 * DD, 256), 256, 0, stream>>>(
      hS, Avul, 16, S_TOT, MAX_STMT, BB * 16, DD);
  gemm(Avul, wiV, fin(IN_GV_BI), nullptr, xiV, nullptr, BB * 16, D3, DD, ACT_NONE);
  zf(hV, (size_t)MPAD * DD); zb(hV_bf, (size_t)MPAD * DD);
  for (int t = 0; t < 16; ++t) {
    gemm(hV_bf, whV, fin(IN_GV_BH), nullptr, gh, nullptr, MPAD, D3, DD, ACT_NONE);
    k_gru_combine<<<cdiv((size_t)BB * DD, 256), 256, 0, stream>>>(xiV, 16, t, gh, hV, hV_bf, BB);
  }

  // --- low proj + LayerNorm + nearest center + high proj ---
  gemm(hV_bf, lowW, fin(IN_LOW_B), nullptr, vlo, nullptr, MPAD, VHD, DD, ACT_NONE);
  k_layernorm_v<<<BB, 256, 0, stream>>>(vlo, fin(IN_LN_G), fin(IN_LN_B), vn);
  zb(clo_bf, (size_t)MPAD * VHD);
  k_nearest<<<BB, 256, 0, stream>>>(vn, fin(IN_CENTERS), clo_bf);
  gemm(clo_bf, highW, fin(IN_HIGH_B), nullptr, chi, nullptr, MPAD, DD, VHD, ACT_NONE);

  // --- T5 encoder ---
  k_build_inp<<<cdiv((size_t)M_T5 * DD, 256), 256, 0, stream>>>(hS, chi, xT);
  k_rel_bias<<<cdiv(LSEQ * LSEQ, 256), 256, 0, stream>>>(fin(IN_REL), biasT);
  for (int l = 0; l < 2; ++l) {
    k_rms<<<M_T5, 256, 0, stream>>>(xT, fin(lln1[l]), h_bf, nullptr, DD);
    gemm(h_bf, qb[l], nullptr, nullptr, qF, nullptr, M_T5, DD, DD, ACT_NONE);
    gemm(h_bf, kb[l], nullptr, nullptr, kF, nullptr, M_T5, DD, DD, ACT_NONE);
    gemm(h_bf, vb[l], nullptr, nullptr, vF, nullptr, M_T5, DD, DD, ACT_NONE);
    zb(ctx_bf, (size_t)M_T5 * DD);
    dim3 ag(LSEQ, NH, BB);
    k_attention<<<ag, 128, 0, stream>>>(qF, kF, vF, biasT, fin(IN_MASK), ctx_bf);
    gemm(ctx_bf, ob[l], nullptr, xT, xT, nullptr, M_T5, DD, DD, ACT_NONE);
    k_rms<<<M_T5, 256, 0, stream>>>(xT, fin(lln2[l]), h_bf, nullptr, DD);
    gemm(h_bf, wib[l], nullptr, nullptr, nullptr, ff_bf, M_T5, FF, DD, ACT_RELU);
    gemm(ff_bf, wob[l], nullptr, xT, xT, nullptr, M_T5, DD, FF, ACT_NONE);
  }
  k_rms<<<M_T5, 256, 0, stream>>>(xT, fin(IN_T5_FINAL_LN), rep_bf, nullptr, DD);

  // --- classifier heads ---
  const int MH = BB * MAX_STMT;  // 1024
  k_gather_rows_b2b<<<cdiv((size_t)MH * DD, 256), 256, 0, stream>>>(
      rep_bf, hf_bf, MAX_STMT, LSEQ, 0, MH, DD);
  gemm(hf_bf, denseW, fin(IN_CLS_DENSE_B), nullptr, xcls, nullptr, MH, DD, DD, ACT_TANH);
  { dim3 g(MH, 1); k_rowdot<<<g, 256, 0, stream>>>(xcls, fin(IN_CLS_OUT_W), fin(IN_CLS_OUT_B), logits, DD, 1); }
  k_stmt_loss<<<1, 256, 0, stream>>>(logits, fin(IN_YSTMT), (float*)d_out);

  // function GRU over h_function (8 seqs, 128 steps)
  gemm(hf_bf, wiC, fin(IN_CLS_GRU_BI), nullptr, xiC, nullptr, MH, D3, DD, ACT_NONE);
  zf(hC, (size_t)MPAD * DD); zb(hC_bf, (size_t)MPAD * DD);
  for (int t = 0; t < MAX_STMT; ++t) {
    gemm(hC_bf, whC, fin(IN_CLS_GRU_BH), nullptr, gh, nullptr, MPAD, D3, DD, ACT_NONE);
    k_gru_combine<<<cdiv((size_t)BB * DD, 256), 256, 0, stream>>>(xiC, MAX_STMT, t, gh, hC, hC_bf, BB);
  }
  gemm(hC_bf, fdenseW, fin(IN_CLS_FDENSE_B), nullptr, fh, nullptr, MPAD, DD, DD, ACT_TANH);
  { dim3 g(BB, 2); k_rowdot<<<g, 256, 0, stream>>>(fh, fin(IN_CLS_FOUT_W), fin(IN_CLS_FOUT_B), flog, DD, 2); }
  k_func_loss<<<1, 32, 0, stream>>>(flog, (const int*)d_in[IN_LABELS], (float*)d_out);
}
